// IVQ_CBM_55954833932989
// MI455X (gfx1250) — matrix operand order repaired and checked by reading, not compiled, not run on hardware
//
#include <hip/hip_runtime.h>
#include <hip/hip_bf16.h>

typedef __attribute__((ext_vector_type(16))) __bf16 v16bf;
typedef __attribute__((ext_vector_type(8)))  float  v8f;
typedef __attribute__((ext_vector_type(2)))  float  v2f;

__device__ __forceinline__ float wave_sum(float v) {
#pragma unroll
  for (int off = 16; off > 0; off >>= 1) v += __shfl_xor(v, off, 32);
  return v;
}

// ---------------- init accumulators ----------------
__global__ void init_acc(float* acc) {
  if (threadIdx.x < 2) acc[threadIdx.x] = 0.0f;
}

// ---------------- pack fp32 weights -> bf16 WMMA B-fragment layout ----------
// Wp[((kt*NT+nt)*32+lane)*16+j] = bf16( W[(kt*32 + 16*(lane>>4) + j)*Ncols + nt*16 + lane%16] )
__global__ void pack_w_bf16(const float* __restrict__ W, __bf16* __restrict__ Wp,
                            int KT, int NT, int Ncols) {
  int t = blockIdx.x * blockDim.x + threadIdx.x;
  int total = KT * NT * 32;
  if (t >= total) return;
  int lane = t & 31;
  int tile = t >> 5;
  int nt = tile % NT;
  int kt = tile / NT;
  int n = nt * 16 + (lane & 15);
  int kbase = kt * 32 + 16 * (lane >> 4);
  __bf16* dst = Wp + (size_t)t * 16;
#pragma unroll
  for (int j = 0; j < 16; j++)
    dst[j] = (__bf16)W[(size_t)(kbase + j) * Ncols + n];
}

// ---------------- distances + softmax + vq loss ----------------
__global__ void dist_soft(const float* __restrict__ X, const float* __restrict__ vq_cb,
                          const float* __restrict__ agg_cb, float* __restrict__ soft,
                          float* __restrict__ vq_acc) {
  __shared__ float shA[7 * 768];
  __shared__ float shV[7 * 768];
  __shared__ float sqA[8], sqV[8];
  int tid = threadIdx.x;
  for (int i = tid; i < 5376; i += 256) { shA[i] = agg_cb[i]; shV[i] = vq_cb[i]; }
  __syncthreads();
  if (tid < 14) {
    int k = tid % 7;
    const float* src = (tid < 7) ? (shA + k * 768) : (shV + k * 768);
    float s = 0.0f;
    for (int e = 0; e < 768; e++) { float v = src[e]; s += v * v; }
    if (tid < 7) sqA[k] = s; else sqV[k] = s;
  }
  __syncthreads();
  int lane = tid & 31, wv = tid >> 5;
  int token = blockIdx.x * 8 + wv;
  if (token >= 256 * 196) return;
  const float* xp = X + (size_t)token * 768;
  float xb[24];
#pragma unroll
  for (int u = 0; u < 24; u++) xb[u] = xp[lane + 32 * u];
  float xsq = 0.0f;
#pragma unroll
  for (int u = 0; u < 24; u++) xsq += xb[u] * xb[u];
  xsq = wave_sum(xsq);
  float dA[7], dV[7];
  for (int k = 0; k < 7; k++) {
    float a = 0.0f, v = 0.0f;
    const float* ca = shA + k * 768 + lane;
    const float* cv = shV + k * 768 + lane;
#pragma unroll
    for (int u = 0; u < 24; u++) { a += xb[u] * ca[32 * u]; v += xb[u] * cv[32 * u]; }
    a = wave_sum(a); v = wave_sum(v);
    dA[k] = xsq + sqA[k] - 2.0f * a;
    dV[k] = xsq + sqV[k] - 2.0f * v;
  }
  float mx = -dA[0];
  for (int k = 1; k < 7; k++) mx = fmaxf(mx, -dA[k]);
  float se = 0.0f, e[7];
  for (int k = 0; k < 7; k++) { e[k] = expf(-dA[k] - mx); se += e[k]; }
#pragma unroll
  for (int k = 0; k < 7; k++)
    if (lane == k) soft[(size_t)token * 7 + k] = e[k] / se;
  float dmin = dV[0];
  for (int k = 1; k < 7; k++) dmin = fminf(dmin, dV[k]);
  if (lane == 0) atomicAdd(vq_acc, dmin);
}

// ---------------- agg[b,k,:] = sum_n soft[b,n,k] * x[b,n,:] ----------------
__global__ void agg_kernel(const float* __restrict__ soft, const float* __restrict__ X,
                           float* __restrict__ agg) {
  __shared__ float sr[196];
  int k = blockIdx.x, b = blockIdx.y;
  int tid = threadIdx.x;
  if (tid < 196) sr[tid] = soft[((size_t)b * 196 + tid) * 7 + k];
  __syncthreads();
  const float* xb = X + (size_t)b * 196 * 768;
#pragma unroll
  for (int c = 0; c < 3; c++) {
    int d = tid + 256 * c;
    float acc = 0.0f;
    for (int n = 0; n < 196; n++) acc += sr[n] * xb[(size_t)n * 768 + d];
    agg[((size_t)b * 7 + k) * 768 + d] = acc;
  }
}

// ---------------- Gram: batched-load 8 K-steps, then 8 WMMAs ----------------
__device__ __forceinline__ void gram_step(const float* __restrict__ pa,
                                          const float* __restrict__ pb,
                                          float ma, float mb, v8f& acc) {
  float2 va[8], vb[8];
#pragma unroll
  for (int u = 0; u < 8; u++) {
    va[u] = *(const float2*)(pa + 4 * u);
    vb[u] = *(const float2*)(pb + 4 * u);
  }
#pragma unroll
  for (int u = 0; u < 8; u++) {
    v2f av; av[0] = va[u].x * ma; av[1] = va[u].y * ma;
    v2f bv; bv[0] = vb[u].x * mb; bv[1] = vb[u].y * mb;
    acc = __builtin_amdgcn_wmma_f32_16x16x4_f32(false, av, false, bv,
                                                (short)0, acc, false, false);
  }
}

__global__ void gram_wmma(const float* __restrict__ X, float* __restrict__ G) {
  int lane = threadIdx.x & 31;
  int wv = threadIdx.x >> 5;
  int tile = blockIdx.x * 8 + wv;
  if (tile >= 169) return;
  int b = blockIdx.y;
  int ti = tile / 13, tj = tile % 13;
  int half = lane >> 4;
  int ra = ti * 16 + (lane & 15);
  int rb = tj * 16 + (lane & 15);
  float ma = ra < 196 ? 1.0f : 0.0f;
  float mb = rb < 196 ? 1.0f : 0.0f;
  int rac = ra < 196 ? ra : 195;
  int rbc = rb < 196 ? rb : 195;
  const float* pa = X + ((size_t)b * 196 + rac) * 768 + 2 * half;
  const float* pb = X + ((size_t)b * 196 + rbc) * 768 + 2 * half;
  v8f acc = {};
  if (ti < 12 && tj < 12) {        // wave-uniform fast path: masks fold to 1.0
    for (int k = 0; k < 768; k += 32) gram_step(pa + k, pb + k, 1.0f, 1.0f, acc);
  } else {
    for (int k = 0; k < 768; k += 32) gram_step(pa + k, pb + k, ma, mb, acc);
  }
#pragma unroll
  for (int r = 0; r < 8; r++) {
    int m = ti * 16 + 8 * half + r;
    int n = tj * 16 + (lane & 15);
    if (m < 196 && n < 196) G[((size_t)b * 196 + m) * 196 + n] = acc[r];
  }
}

// ---------------- batched parallel Jacobi eigensolve in 157KB LDS -----------
__global__ void jacobi_kernel(const float* __restrict__ Gin, float* __restrict__ rank_acc) {
  extern __shared__ float sh[];
  const int LDG = 197;
  float* G  = sh;
  float* cs = sh + 196 * LDG;      // 196 floats (98 pairs * {c,s})
  int*   pq = (int*)(cs + 196);    // 196 ints
  float* sv  = (float*)(pq + 196); // 196
  float* srt = sv + 196;           // 196
  int b = blockIdx.x;
  int tid = threadIdx.x;
  const float* src = Gin + (size_t)b * 196 * 196;
  for (int i = tid; i < 196 * 196; i += blockDim.x)
    G[(i / 196) * LDG + (i % 196)] = src[i];
  __syncthreads();
  for (int sweep = 0; sweep < 8; ++sweep) {
    for (int rr = 0; rr < 195; ++rr) {
      if (tid < 98) {
        int p, q;
        if (tid == 0) { p = 195; q = rr; }
        else { p = (rr + tid) % 195; q = (rr + 195 - tid) % 195; }
        float app = G[p * LDG + p], aqq = G[q * LDG + q], apq = G[p * LDG + q];
        float c = 1.0f, s = 0.0f;
        if (fabsf(apq) > 1e-12f) {
          float th = (aqq - app) / (2.0f * apq);
          float t = 1.0f / (fabsf(th) + sqrtf(th * th + 1.0f));
          t = th >= 0.0f ? t : -t;
          c = rsqrtf(t * t + 1.0f);
          s = t * c;
        }
        cs[2 * tid] = c; cs[2 * tid + 1] = s;
        pq[2 * tid] = p; pq[2 * tid + 1] = q;
      }
      __syncthreads();
      for (int item = tid; item < 98 * 196; item += blockDim.x) {
        int i = item / 196, j = item % 196;
        int p = pq[2 * i], q = pq[2 * i + 1];
        float c = cs[2 * i], s = cs[2 * i + 1];
        float gp = G[p * LDG + j], gq = G[q * LDG + j];
        G[p * LDG + j] = c * gp - s * gq;
        G[q * LDG + j] = s * gp + c * gq;
      }
      __syncthreads();
      for (int item = tid; item < 98 * 196; item += blockDim.x) {
        int i = item / 196, j = item % 196;
        int p = pq[2 * i], q = pq[2 * i + 1];
        float c = cs[2 * i], s = cs[2 * i + 1];
        float gp = G[j * LDG + p], gq = G[j * LDG + q];
        G[j * LDG + p] = c * gp - s * gq;
        G[j * LDG + q] = s * gp + c * gq;
      }
      __syncthreads();
    }
  }
  if (tid < 196) sv[tid] = sqrtf(fmaxf(G[tid * LDG + tid], 0.0f));
  __syncthreads();
  if (tid < 196) {
    float mine = sv[tid];
    int rank = 0;
    for (int j = 0; j < 196; j++) {
      float o = sv[j];
      rank += (o > mine) || (o == mine && j < tid);
    }
    srt[rank] = mine;
  }
  __syncthreads();
  if (tid == 0) {
    float tot = 0.0f;
    for (int i = 0; i < 196; i++) tot += srt[i];
    float run = 0.0f; int cnt = 0;
    for (int i = 0; i < 196; i++) { run += srt[i]; if (run <= 0.99f * tot) cnt++; }
    atomicAdd(rank_acc, (float)cnt);
  }
}

// ---------------- bf16 WMMA GEMM, 2x2 register-blocked + pipelined ----------
__device__ __forceinline__ v16bf load_a_frag(const float* __restrict__ ap) {
  float4 u0 = *(const float4*)(ap);
  float4 u1 = *(const float4*)(ap + 4);
  float4 u2 = *(const float4*)(ap + 8);
  float4 u3 = *(const float4*)(ap + 12);
  v16bf a;
  a[0]  = (__bf16)u0.x; a[1]  = (__bf16)u0.y; a[2]  = (__bf16)u0.z; a[3]  = (__bf16)u0.w;
  a[4]  = (__bf16)u1.x; a[5]  = (__bf16)u1.y; a[6]  = (__bf16)u1.z; a[7]  = (__bf16)u1.w;
  a[8]  = (__bf16)u2.x; a[9]  = (__bf16)u2.y; a[10] = (__bf16)u2.z; a[11] = (__bf16)u2.w;
  a[12] = (__bf16)u3.x; a[13] = (__bf16)u3.y; a[14] = (__bf16)u3.z; a[15] = (__bf16)u3.w;
  return a;
}

// mode 0: bias + exact GELU; mode 1: bias; mode 2: none
__global__ void gemm_bf16(const float* __restrict__ A, const __bf16* __restrict__ Wp,
                          const float* __restrict__ bias, float* __restrict__ Out,
                          int Mt2, int Nt2, int Ntiles, int KT, int ldA, int ldOut,
                          int mode) {
  int lane = threadIdx.x & 31;
  int wv = threadIdx.x >> 5;
  int tile = blockIdx.x * 8 + wv;
  if (tile >= Mt2 * Nt2) return;
  int n2 = tile % Nt2;
  int m2 = tile / Nt2;
  int half = lane >> 4;
  const float* arow0 = A + ((size_t)(m2 * 32 + (lane & 15))) * ldA + 16 * half;
  const float* arow1 = arow0 + (size_t)16 * ldA;
  size_t bstride = (size_t)Ntiles * 512;  // elements per kt step
  const __bf16* bp0 = Wp + (((size_t)(n2 * 2 + 0)) * 32 + lane) * 16;
  const __bf16* bp1 = Wp + (((size_t)(n2 * 2 + 1)) * 32 + lane) * 16;
  v8f acc00 = {}, acc01 = {}, acc10 = {}, acc11 = {};
  v16bf a0c = load_a_frag(arow0);
  v16bf a1c = load_a_frag(arow1);
  v16bf b0c = *(const v16bf*)(bp0);
  v16bf b1c = *(const v16bf*)(bp1);
  for (int kt = 0; kt < KT; ++kt) {
    int kn = (kt + 1 < KT) ? (kt + 1) : kt;
    v16bf a0n = load_a_frag(arow0 + kn * 32);
    v16bf a1n = load_a_frag(arow1 + kn * 32);
    v16bf b0n = *(const v16bf*)(bp0 + (size_t)kn * bstride);
    v16bf b1n = *(const v16bf*)(bp1 + (size_t)kn * bstride);
    acc00 = __builtin_amdgcn_wmma_f32_16x16x32_bf16(false, a0c, false, b0c,
                                                    (short)0, acc00, false, false);
    acc01 = __builtin_amdgcn_wmma_f32_16x16x32_bf16(false, a0c, false, b1c,
                                                    (short)0, acc01, false, false);
    acc10 = __builtin_amdgcn_wmma_f32_16x16x32_bf16(false, a1c, false, b0c,
                                                    (short)0, acc10, false, false);
    acc11 = __builtin_amdgcn_wmma_f32_16x16x32_bf16(false, a1c, false, b1c,
                                                    (short)0, acc11, false, false);
    a0c = a0n; a1c = a1n; b0c = b0n; b1c = b1n;
  }
  int nb0 = (n2 * 2) * 16 + (lane & 15);
  int nb1 = nb0 + 16;
  float bv0 = bias ? bias[nb0] : 0.0f;
  float bv1 = bias ? bias[nb1] : 0.0f;
#pragma unroll
  for (int r = 0; r < 8; r++) {
    int row0 = m2 * 32 + 8 * half + r;
    int row1 = row0 + 16;
    float v00 = acc00[r] + bv0, v01 = acc01[r] + bv1;
    float v10 = acc10[r] + bv0, v11 = acc11[r] + bv1;
    if (mode == 0) {
      v00 = 0.5f * v00 * (1.0f + erff(v00 * 0.70710678118654752f));
      v01 = 0.5f * v01 * (1.0f + erff(v01 * 0.70710678118654752f));
      v10 = 0.5f * v10 * (1.0f + erff(v10 * 0.70710678118654752f));
      v11 = 0.5f * v11 * (1.0f + erff(v11 * 0.70710678118654752f));
    }
    Out[(size_t)row0 * ldOut + nb0] = v00;
    Out[(size_t)row0 * ldOut + nb1] = v01;
    Out[(size_t)row1 * ldOut + nb0] = v10;
    Out[(size_t)row1 * ldOut + nb1] = v11;
  }
}

// ---------------- LayerNorm over last dim (768), in place -------------------
__global__ void ln_kernel(float* __restrict__ H, const float* __restrict__ gamma,
                          const float* __restrict__ beta) {
  __shared__ float red[256];
  int row = blockIdx.x, tid = threadIdx.x;
  float* hp = H + (size_t)row * 768;
  float v0 = hp[tid], v1 = hp[tid + 256], v2 = hp[tid + 512];
  red[tid] = v0 + v1 + v2;
  __syncthreads();
  for (int off = 128; off > 0; off >>= 1) {
    if (tid < off) red[tid] += red[tid + off];
    __syncthreads();
  }
  float mu = red[0] / 768.0f;
  __syncthreads();
  float d0 = v0 - mu, d1 = v1 - mu, d2 = v2 - mu;
  red[tid] = d0 * d0 + d1 * d1 + d2 * d2;
  __syncthreads();
  for (int off = 128; off > 0; off >>= 1) {
    if (tid < off) red[tid] += red[tid + off];
    __syncthreads();
  }
  float rstd = rsqrtf(red[0] / 768.0f + 1e-5f);
  hp[tid]       = d0 * rstd * gamma[tid]       + beta[tid];
  hp[tid + 256] = d1 * rstd * gamma[tid + 256] + beta[tid + 256];
  hp[tid + 512] = d2 * rstd * gamma[tid + 512] + beta[tid + 512];
}

// ---------------- token L2 norms ----------------
__global__ void toknorm(const float* __restrict__ tok, float* __restrict__ norms) {
  int lane = threadIdx.x & 31, wv = threadIdx.x >> 5;
  int row = blockIdx.x * 8 + wv;
  if (row >= 1792) return;
  const float* tp = tok + (size_t)row * 512;
  float s = 0.0f;
#pragma unroll
  for (int u = 0; u < 16; u++) { float v = tp[lane + 32 * u]; s += v * v; }
  s = wave_sum(s);
  if (lane == 0) norms[row] = sqrtf(s);
}

// ---------------- grouped concept logits ----------------
__global__ void logits_kernel(const float* __restrict__ tok, const float* __restrict__ norms,
                              const float* __restrict__ ce, const float* __restrict__ lscale,
                              float* __restrict__ out_il) {
  int lane = threadIdx.x & 31, wv = threadIdx.x >> 5;
  int pair = blockIdx.x * 8 + wv;
  if (pair >= 256 * 34) return;
  int b = pair / 34, c = pair % 34;
  int g = c / 5; if (g > 6) g = 6;
  const float* tp = tok + ((size_t)b * 7 + g) * 512;
  const float* cp = ce + (size_t)c * 512;
  float s = 0.0f;
#pragma unroll
  for (int u = 0; u < 16; u++) s += tp[lane + 32 * u] * cp[lane + 32 * u];
  s = wave_sum(s);
  if (lane == 0) out_il[(size_t)b * 34 + c] = lscale[0] * s / norms[b * 7 + g];
}

// ---------------- classifier head + scalar outputs ----------------
__global__ void final_kernel(const float* __restrict__ il, const float* __restrict__ clsW,
                             const float* __restrict__ clsB, const float* __restrict__ acc,
                             float* __restrict__ out) {
  int t = blockIdx.x * blockDim.x + threadIdx.x;
  if (t < 1792) {
    int b = t / 7, j = t % 7;
    float s = clsB[j];
    for (int c = 0; c < 34; c++) s += il[b * 34 + c] * clsW[c * 7 + j];
    out[t] = s;
  }
  if (t == 0) {
    out[1792 + 8704]     = 2.0f * acc[0] / (256.0f * 196.0f * 768.0f);
    out[1792 + 8704 + 1] = acc[1] / 256.0f;
  }
}

extern "C" void kernel_launch(void* const* d_in, const int* in_sizes, int n_in,
                              void* d_out, int out_size, void* d_ws, size_t ws_size,
                              hipStream_t stream) {
  (void)in_sizes; (void)n_in; (void)out_size; (void)ws_size;
  const float* X      = (const float*)d_in[0];   // (256,196,768)
  const float* vq_cb  = (const float*)d_in[1];   // (7,768)
  const float* agg_cb = (const float*)d_in[2];   // (7,768)
  const float* W1     = (const float*)d_in[3];   // (768,3072)
  const float* b1     = (const float*)d_in[4];
  const float* W2     = (const float*)d_in[5];   // (3072,768)
  const float* b2     = (const float*)d_in[6];
  const float* gamma  = (const float*)d_in[7];
  const float* beta   = (const float*)d_in[8];
  const float* Wproj  = (const float*)d_in[9];   // (768,512)
  const float* ce     = (const float*)d_in[10];  // (34,512)
  const float* clsW   = (const float*)d_in[11];  // (34,7)
  const float* clsB   = (const float*)d_in[12];
  const float* lscale = (const float*)d_in[13];
  float* out = (float*)d_out;

  size_t off = 0;
  auto alloc = [&](size_t bytes) -> char* {
    char* p = (char*)d_ws + off;
    off += (bytes + 255) & ~(size_t)255;
    return p;
  };
  float*  acc   = (float*)alloc(2 * 4);
  __bf16* W1p   = (__bf16*)alloc((size_t)768 * 3072 * 2);
  __bf16* W2p   = (__bf16*)alloc((size_t)3072 * 768 * 2);
  __bf16* Wpp   = (__bf16*)alloc((size_t)768 * 512 * 2);
  float*  soft  = (float*)alloc((size_t)50176 * 7 * 4);
  float*  aggb  = (float*)alloc((size_t)1792 * 768 * 4);
  float*  h2    = (float*)alloc((size_t)1792 * 768 * 4);
  float*  tokb  = (float*)alloc((size_t)1792 * 512 * 4);
  float*  norms = (float*)alloc((size_t)1792 * 4);
  float*  Gbuf  = (float*)alloc((size_t)256 * 196 * 196 * 4);
  float*  h1    = Gbuf;  // reused after jacobi consumes G (stream-ordered)

  init_acc<<<1, 64, 0, stream>>>(acc);
  pack_w_bf16<<<(24 * 192 * 32 + 255) / 256, 256, 0, stream>>>(W1, W1p, 24, 192, 3072);
  pack_w_bf16<<<(96 * 48 * 32 + 255) / 256, 256, 0, stream>>>(W2, W2p, 96, 48, 768);
  pack_w_bf16<<<(24 * 32 * 32 + 255) / 256, 256, 0, stream>>>(Wproj, Wpp, 24, 32, 512);
  dist_soft<<<6272, 256, 0, stream>>>(X, vq_cb, agg_cb, soft, acc + 0);
  agg_kernel<<<dim3(7, 256), 256, 0, stream>>>(soft, X, aggb);
  gram_wmma<<<dim3(22, 256), 256, 0, stream>>>(X, Gbuf);
  const int jacobi_lds = (196 * 197 + 196 + 196 + 196 + 196) * 4;  // 157584 B
  hipFuncSetAttribute(reinterpret_cast<const void*>(jacobi_kernel),
                      hipFuncAttributeMaxDynamicSharedMemorySize, jacobi_lds);
  jacobi_kernel<<<256, 1024, jacobi_lds, stream>>>(Gbuf, acc + 1);
  // 2x2-blocked GEMMs: grid = (Mt2*Nt2)/8 waves-blocks
  gemm_bf16<<<(56 * 96) / 8, 256, 0, stream>>>(aggb, W1p, b1, h1, 56, 96, 192, 24, 768, 3072, 0);
  gemm_bf16<<<(56 * 24) / 8, 256, 0, stream>>>(h1, W2p, b2, h2, 56, 24, 48, 96, 3072, 768, 1);
  ln_kernel<<<1792, 256, 0, stream>>>(h2, gamma, beta);
  gemm_bf16<<<(56 * 16) / 8, 256, 0, stream>>>(h2, Wpp, nullptr, tokb, 56, 16, 32, 24, 768, 512, 2);
  toknorm<<<224, 256, 0, stream>>>(tokb, norms);
  logits_kernel<<<1088, 256, 0, stream>>>(tokb, norms, ce, lscale, out + 1792);
  final_kernel<<<7, 256, 0, stream>>>(out + 1792, clsW, clsB, acc, out);
}
